// RegionAwareTokenFusionV2_32942399160400
// MI455X (gfx1250) — matrix operand order repaired and checked by reading, not compiled, not run on hardware
//
#include <hip/hip_runtime.h>
#include <hip/hip_bf16.h>
#include <cmath>

typedef __attribute__((ext_vector_type(16))) _Float16 v16h;
typedef __attribute__((ext_vector_type(8)))  _Float16 v8h;
typedef __attribute__((ext_vector_type(8)))  float    v8f;

#define BATCH 64
#define CH    768
#define TOK   576
#define TP1   577
#define NH    8
#define HD    96
#define NKV   1536            // k (768) and v (768) fused along N
#define G1DIM 192
#define EPSLN 1e-6f

// ---------------- block-wide reductions (blockDim.x == 256) ----------------
__device__ __forceinline__ float block_reduce_sum(float v, float* sm) {
  int tid = threadIdx.x;
  sm[tid] = v; __syncthreads();
  for (int s = 128; s > 0; s >>= 1) { if (tid < s) sm[tid] += sm[tid + s]; __syncthreads(); }
  float r = sm[0]; __syncthreads();
  return r;
}
__device__ __forceinline__ float block_reduce_max(float v, float* sm) {
  int tid = threadIdx.x;
  sm[tid] = v; __syncthreads();
  for (int s = 128; s > 0; s >>= 1) { if (tid < s) sm[tid] = fmaxf(sm[tid], sm[tid + s]); __syncthreads(); }
  float r = sm[0]; __syncthreads();
  return r;
}

// ---------------- 1) pre-LN over spatial dims, + cls0 = mean_t(tokens) ------
__global__ void k_preln(const float* __restrict__ feat, const float* __restrict__ w,
                        const float* __restrict__ bb, float* __restrict__ x,
                        float* __restrict__ cls0) {
  __shared__ float sm[256];
  int row = blockIdx.x;                 // b*CH + c
  int c = row % CH;
  const float* fp = feat + (size_t)row * TOK;
  float s = 0.f, s2 = 0.f;
  for (int t = threadIdx.x; t < TOK; t += 256) { float f = fp[t]; s += f; s2 += f * f; }
  float sum   = block_reduce_sum(s,  sm);
  float sumsq = block_reduce_sum(s2, sm);
  float u    = sum * (1.0f / TOK);
  float var  = sumsq * (1.0f / TOK) - u * u;
  float rstd = rsqrtf(var + EPSLN);
  float wc = w[c], bc = bb[c];
  float xs = 0.f;
  float* xp = x + (size_t)row * TOK;
  for (int t = threadIdx.x; t < TOK; t += 256) {
    float xv = wc * ((fp[t] - u) * rstd) + bc;
    xp[t] = xv; xs += xv;
  }
  float xsum = block_reduce_sum(xs, sm);
  if (threadIdx.x == 0) cls0[row] = xsum * (1.0f / TOK);
}

// ---------------- 2) pack xx = [cls0 ; tokens] + pos, f16, (B,577,768) ------
__global__ void k_pack_cls(const float* __restrict__ cls0, const float* __restrict__ pos,
                           _Float16* __restrict__ xx) {
  int i = blockIdx.x * 256 + threadIdx.x;        // < BATCH*CH
  int b = i / CH, c = i % CH;
  xx[(size_t)b * TP1 * CH + c] = (_Float16)(cls0[i] + pos[c]);
}

__global__ void k_pack_tok(const float* __restrict__ x, const float* __restrict__ pos,
                           _Float16* __restrict__ xx) {
  __shared__ float tile[32][33];
  int b = blockIdx.z, c0 = blockIdx.x * 32, t0 = blockIdx.y * 32;
  int tx = threadIdx.x & 31, ty = threadIdx.x >> 5;     // 32 x 8
#pragma unroll
  for (int i = 0; i < 4; ++i) {
    int c = c0 + ty + i * 8;
    tile[ty + i * 8][tx] = x[(size_t)(b * CH + c) * TOK + t0 + tx];   // coalesced in t
  }
  __syncthreads();
#pragma unroll
  for (int i = 0; i < 4; ++i) {
    int t = t0 + ty + i * 8;
    int c = c0 + tx;
    xx[(size_t)(b * TP1 + 1 + t) * CH + c] =
        (_Float16)(tile[tx][ty + i * 8] + pos[(size_t)(1 + t) * CH + c]);  // coalesced in c
  }
}

// ---------------- 3) convert [Wk;Wv] to f16 (1536 x 768, row-major N,K) ----
__global__ void k_cvt_wkv(const float* __restrict__ Wk, const float* __restrict__ Wv,
                          _Float16* __restrict__ Wkv) {
  int i = blockIdx.x * 256 + threadIdx.x;        // < NKV*CH
  int n = i / CH, c = i % CH;
  float v = (n < CH) ? Wk[i] : Wv[(size_t)(n - CH) * CH + c];
  Wkv[i] = (_Float16)v;
}

// ---------------- 4) fused K/V GEMM: (36928 x 768) f16 @ (768 x 1536) -------
// One wave computes a 16(M) x 64(N) strip: 4 accumulators, A-fragment reuse 4x.
// A frag (16x32 f16): lane group g holds contiguous K-run kk+16g..+15.
// B frag (32x16 f16): lane holds contiguous K-run kk+16g..+15 of column n.
__global__ void __launch_bounds__(256) k_gemm_kv(const _Float16* __restrict__ A,
                                                 const _Float16* __restrict__ Bm,
                                                 const float* __restrict__ bk,
                                                 const float* __restrict__ bv,
                                                 _Float16* __restrict__ kvout) {
  int lane = threadIdx.x & 31;
  int wave = threadIdx.x >> 5;                   // 0..7
  int g    = lane >> 4;                          // half-wave group
  int l15  = lane & 15;
  int mt   = blockIdx.y;                         // 16-row M tile
  int nb   = blockIdx.x * 512 + wave * 64;       // wave's N base

  const _Float16* arow = A + (size_t)(mt * 16 + l15) * CH;
  v8f acc[4] = {};
  for (int kk = 0; kk < CH; kk += 32) {
    v8h alo = *(const v8h*)(arow + kk + 16 * g);
    v8h ahi = *(const v8h*)(arow + kk + 16 * g + 8);
    v16h a = __builtin_shufflevector(alo, ahi, 0, 1, 2, 3, 4, 5, 6, 7,
                                     8, 9, 10, 11, 12, 13, 14, 15);
    const _Float16* bbase = Bm + kk + 16 * g;
#pragma unroll
    for (int j = 0; j < 4; ++j) {
      v16h bfr = *(const v16h*)(bbase + (size_t)(nb + j * 16 + l15) * CH);
      acc[j] = __builtin_amdgcn_wmma_f32_16x16x32_f16(
          /*neg_a=*/false, a, /*neg_b=*/false, bfr,
          /*c_mod=*/(short)0, acc[j], /*reuse_a=*/false, /*reuse_b=*/false);
    }
  }
  // D layout: element r of acc -> M = mt*16 + r + 8g, N = nb + j*16 + l15
#pragma unroll
  for (int j = 0; j < 4; ++j) {
    int nn = nb + j * 16 + l15;
    float bias = (nn < CH) ? bk[nn] : bv[nn - CH];
    _Float16* op = kvout + (size_t)(mt * 16 + 8 * g) * NKV + nn;
#pragma unroll
    for (int r = 0; r < 8; ++r)
      op[(size_t)r * NKV] = (_Float16)(acc[j][r] + bias);
  }
}

// ---------------- 5) q projection (64 rows only, f32) -----------------------
__global__ void k_qproj(const float* __restrict__ cls0, const float* __restrict__ pos,
                        const float* __restrict__ Wq, const float* __restrict__ bq,
                        float* __restrict__ q) {
  __shared__ float sm[256];
  int n = blockIdx.x % CH, b = blockIdx.x / CH;
  const float* wr = Wq + (size_t)n * CH;
  const float* cp = cls0 + (size_t)b * CH;
  float s = 0.f;
  for (int c = threadIdx.x; c < CH; c += 256) s += (cp[c] + pos[c]) * wr[c];
  float tot = block_reduce_sum(s, sm);
  if (threadIdx.x == 0) q[(size_t)b * CH + n] = tot + bq[n];
}

// ---------------- 6) attention logits, softmax, pool ------------------------
__global__ void k_logits(const float* __restrict__ q, const _Float16* __restrict__ kv,
                         float* __restrict__ lg) {
  int i = blockIdx.x * 256 + threadIdx.x;
  if (i >= BATCH * NH * TP1) return;
  int t = i % TP1; int bh = i / TP1; int h = bh % NH; int b = bh / NH;
  const float*    qp = q  + (size_t)b * CH + h * HD;
  const _Float16* kp = kv + (size_t)(b * TP1 + t) * NKV + h * HD;
  float s = 0.f;
#pragma unroll 8
  for (int d = 0; d < HD; ++d) s += qp[d] * (float)kp[d];
  lg[i] = s * 0.10206207261596577f;       // 1/sqrt(96)
}

__global__ void k_softmax(float* __restrict__ lg) {
  __shared__ float sm[256];
  __shared__ float pr[TP1];
  float* row = lg + (size_t)blockIdx.x * TP1;     // over B*NH
  float mx = -1e30f;
  for (int t = threadIdx.x; t < TP1; t += 256) { float v = row[t]; pr[t] = v; mx = fmaxf(mx, v); }
  mx = block_reduce_max(mx, sm);
  float s = 0.f;
  for (int t = threadIdx.x; t < TP1; t += 256) { float e = expf(pr[t] - mx); pr[t] = e; s += e; }
  s = block_reduce_sum(s, sm);
  float inv = 1.0f / s;
  for (int t = threadIdx.x; t < TP1; t += 256) row[t] = pr[t] * inv;
}

__global__ void k_pooled(const float* __restrict__ p, const _Float16* __restrict__ kv,
                         float* __restrict__ pooled) {
  int i = blockIdx.x * 256 + threadIdx.x;         // < B*NH*HD
  int d = i % HD; int bh = i / HD; int h = bh % NH; int b = bh / NH;
  const float*    pr = p  + (size_t)bh * TP1;
  const _Float16* vp = kv + (size_t)b * TP1 * NKV + CH + h * HD + d;
  float s = 0.f;
  for (int t = 0; t < TP1; ++t) s += pr[t] * (float)vp[(size_t)t * NKV];
  pooled[(size_t)b * CH + h * HD + d] = s;
}

// ---------------- 7) attn_sp = head-mean, max-normalized --------------------
__global__ void k_attnsp(const float* __restrict__ p, float* __restrict__ asp,
                         float* __restrict__ out_asp) {
  __shared__ float sm[256];
  __shared__ float sp[TOK];
  int b = blockIdx.x;
  float mx = -1e30f;
  for (int t = threadIdx.x; t < TOK; t += 256) {
    float s = 0.f;
#pragma unroll
    for (int h = 0; h < NH; ++h) s += p[(size_t)(b * NH + h) * TP1 + 1 + t];
    s *= (1.0f / NH);
    sp[t] = s; mx = fmaxf(mx, s);
  }
  mx = block_reduce_max(mx, sm);
  float inv = 1.0f / (mx + 1e-6f);
  for (int t = threadIdx.x; t < TOK; t += 256) {
    float v = sp[t] * inv;
    asp[(size_t)b * TOK + t] = v;
    out_asp[(size_t)b * TOK + t] = v;
  }
}

// ---------------- 8) output projection of pooled ----------------------------
__global__ void k_oproj(const float* __restrict__ pooled, const float* __restrict__ Wo,
                        const float* __restrict__ bo, float* __restrict__ o) {
  __shared__ float sm[256];
  int n = blockIdx.x % CH, b = blockIdx.x / CH;
  const float* wr = Wo + (size_t)n * CH;
  const float* pp = pooled + (size_t)b * CH;
  float s = 0.f;
  for (int c = threadIdx.x; c < CH; c += 256) s += pp[c] * wr[c];
  float tot = block_reduce_sum(s, sm);
  if (threadIdx.x == 0) o[(size_t)b * CH + n] = tot + bo[n];
}

// ---------------- 9) saliency + top-k + refine/fuse -------------------------
__global__ void k_sal(const float* __restrict__ x, float* __restrict__ sal) {
  int i = blockIdx.x * 256 + threadIdx.x;   // < B*TOK, t fast -> coalesced
  int t = i % TOK, b = i / TOK;
  const float* xp = x + (size_t)b * CH * TOK + t;
  float s = 0.f;
  for (int c = 0; c < CH; ++c) { float v = xp[(size_t)c * TOK]; s += v * v; }
  sal[i] = s * (1.0f / CH);
}

__global__ void k_topk(const float* __restrict__ sal, int* __restrict__ idx, int K) {
  __shared__ float s[TOK];
  __shared__ float bv[256];
  __shared__ int   bi[256];
  int b = blockIdx.x, tid = threadIdx.x;
  for (int t = tid; t < TOK; t += 256) s[t] = sal[(size_t)b * TOK + t];
  __syncthreads();
  for (int k = 0; k < K; ++k) {
    float best = -1e30f; int bidx = TOK;
    for (int t = tid; t < TOK; t += 256) {
      float v = s[t];
      if (v > best || (v == best && t < bidx)) { best = v; bidx = t; }
    }
    bv[tid] = best; bi[tid] = bidx; __syncthreads();
    for (int st = 128; st > 0; st >>= 1) {
      if (tid < st) {
        float ov = bv[tid + st]; int oi = bi[tid + st];
        if (ov > bv[tid] || (ov == bv[tid] && oi < bi[tid])) { bv[tid] = ov; bi[tid] = oi; }
      }
      __syncthreads();
    }
    if (tid == 0) { idx[(size_t)b * K + k] = bi[0]; s[bi[0]] = -1e30f; }
    __syncthreads();
  }
}

__global__ void k_vfused(const float* __restrict__ x, const int* __restrict__ idx,
                         const float* __restrict__ o, const float* __restrict__ cls0,
                         float* __restrict__ vf, int K) {
  int i = blockIdx.x * 256 + threadIdx.x;   // < B*CH
  int b = i / CH;
  const float* xp = x + (size_t)i * TOK;
  float s = 0.f;
  for (int j = 0; j < K; ++j) s += xp[idx[(size_t)b * K + j]];
  float refine = s / (float)K;
  float vfg = 0.8f * o[i] + 0.2f * refine;
  vf[i] = 0.5f * cls0[i] + 0.5f * vfg;      // GAMMA = 0.5
}

// ---------------- 10) gate MLP ----------------------------------------------
__global__ void k_g1(const float* __restrict__ vf, const float* __restrict__ Wg1,
                     float* __restrict__ h1) {
  __shared__ float sm[256];
  int j = blockIdx.x % G1DIM, b = blockIdx.x / G1DIM;
  const float* wr = Wg1 + (size_t)j * CH;
  const float* vp = vf + (size_t)b * CH;
  float s = 0.f;
  for (int c = threadIdx.x; c < CH; c += 256) s += vp[c] * wr[c];
  s = block_reduce_sum(s, sm);
  if (threadIdx.x == 0) h1[(size_t)b * G1DIM + j] = fmaxf(s, 0.0f);
}

__global__ void k_g2(const float* __restrict__ h1, const float* __restrict__ Wg2,
                     const float* __restrict__ bg2, float* __restrict__ gate) {
  __shared__ float sm[256];
  int n = blockIdx.x % CH, b = blockIdx.x / CH;
  const float* wr = Wg2 + (size_t)n * G1DIM;
  const float* hp = h1 + (size_t)b * G1DIM;
  float s = (threadIdx.x < G1DIM) ? hp[threadIdx.x] * wr[threadIdx.x] : 0.f;
  s = block_reduce_sum(s, sm);
  if (threadIdx.x == 0) {
    float z = s + bg2[n];
    gate[(size_t)b * CH + n] = 1.0f / (1.0f + expf(-z));
  }
}

// ---------------- 11) fused * gate * (1+attn_sp), post-LN, blend ------------
__global__ void k_final(const float* __restrict__ feat, const float* __restrict__ x,
                        const float* __restrict__ gate, const float* __restrict__ asp,
                        const float* __restrict__ pw, const float* __restrict__ pb,
                        float* __restrict__ out, float alpha) {
  __shared__ float sm[256];
  __shared__ float fv[TOK];
  int row = blockIdx.x;                 // b*CH + c
  int c = row % CH, b = row / CH;
  float g = gate[row];
  const float* xp = x + (size_t)row * TOK;
  const float* ap = asp + (size_t)b * TOK;
  float s = 0.f, s2 = 0.f;
  for (int t = threadIdx.x; t < TOK; t += 256) {
    float v = xp[t] * g * (1.0f + ap[t]);
    fv[t] = v; s += v; s2 += v * v;
  }
  float sum   = block_reduce_sum(s,  sm);
  float sumsq = block_reduce_sum(s2, sm);
  float u    = sum * (1.0f / TOK);
  float var  = sumsq * (1.0f / TOK) - u * u;
  float rstd = rsqrtf(var + EPSLN);
  float wc = pw[c], bc = pb[c];
  const float* fp = feat + (size_t)row * TOK;
  float* op = out + (size_t)row * TOK;
  for (int t = threadIdx.x; t < TOK; t += 256) {
    float ln = wc * ((fv[t] - u) * rstd) + bc;
    float f0 = fp[t];
    op[t] = f0 + alpha * (ln - f0);
  }
}

// ---------------- host launcher ---------------------------------------------
extern "C" void kernel_launch(void* const* d_in, const int* in_sizes, int n_in,
                              void* d_out, int out_size, void* d_ws, size_t ws_size,
                              hipStream_t stream) {
  (void)in_sizes; (void)n_in; (void)out_size; (void)ws_size;
  const float* feat   = (const float*)d_in[0];
  const float* pre_w  = (const float*)d_in[1];
  const float* pre_b  = (const float*)d_in[2];
  const float* post_w = (const float*)d_in[3];
  const float* post_b = (const float*)d_in[4];
  const float* pos    = (const float*)d_in[5];
  const float* Wq     = (const float*)d_in[6];
  const float* bq     = (const float*)d_in[7];
  const float* Wk     = (const float*)d_in[8];
  const float* bk     = (const float*)d_in[9];
  const float* Wv     = (const float*)d_in[10];
  const float* bv     = (const float*)d_in[11];
  const float* Wo     = (const float*)d_in[12];
  const float* bo     = (const float*)d_in[13];
  const float* Wg1    = (const float*)d_in[14];
  const float* Wg2    = (const float*)d_in[15];
  const float* bg2    = (const float*)d_in[16];

  // schedule constants (STEP=1): ALPHA ~= 3.838e-7, K = 28
  const double PI = 3.14159265358979323846;
  double ratio = 0.5 * (1.0 - cos(PI * 1.0 / 1500.0));
  float alpha = (float)(ratio * 0.35);
  int K = (int)((double)TOK * (0.05 + 0.10 * ratio));
  if (K < 1) K = 1;

  // workspace carve-out (256B aligned)
  char* ws = (char*)d_ws;
  size_t off = 0;
  auto carve = [&](size_t bytes) -> void* {
    void* p = ws + off;
    off += (bytes + 255) & ~(size_t)255;
    return p;
  };
  float*    x      = (float*)   carve((size_t)BATCH * CH * TOK * 4);   // pre-LN output
  float*    cls0   = (float*)   carve((size_t)BATCH * CH * 4);
  _Float16* xx     = (_Float16*)carve((size_t)BATCH * TP1 * CH * 2);   // tokens+pos, f16
  _Float16* Wkv    = (_Float16*)carve((size_t)NKV * CH * 2);
  _Float16* kv     = (_Float16*)carve((size_t)BATCH * TP1 * NKV * 2);  // k|v, f16
  float*    q      = (float*)   carve((size_t)BATCH * CH * 4);
  float*    lg     = (float*)   carve((size_t)BATCH * NH * TP1 * 4);   // logits -> probs
  float*    pooled = (float*)   carve((size_t)BATCH * CH * 4);
  float*    o      = (float*)   carve((size_t)BATCH * CH * 4);
  float*    asp    = (float*)   carve((size_t)BATCH * TOK * 4);
  float*    sal    = (float*)   carve((size_t)BATCH * TOK * 4);
  int*      idxb   = (int*)     carve((size_t)BATCH * 64 * 4);
  float*    vf     = (float*)   carve((size_t)BATCH * CH * 4);
  float*    h1     = (float*)   carve((size_t)BATCH * G1DIM * 4);
  float*    gate   = (float*)   carve((size_t)BATCH * CH * 4);

  float* out     = (float*)d_out;
  float* out_asp = out + (size_t)BATCH * CH * TOK;

  k_preln   <<<BATCH * CH, 256, 0, stream>>>(feat, pre_w, pre_b, x, cls0);
  k_pack_cls<<<(BATCH * CH) / 256, 256, 0, stream>>>(cls0, pos, xx);
  k_pack_tok<<<dim3(CH / 32, TOK / 32, BATCH), 256, 0, stream>>>(x, pos, xx);
  k_cvt_wkv <<<(NKV * CH) / 256, 256, 0, stream>>>(Wk, Wv, Wkv);
  k_gemm_kv <<<dim3(NKV / 512, (BATCH * TP1) / 16), 256, 0, stream>>>(xx, Wkv, bk, bv, kv);
  k_qproj   <<<BATCH * CH, 256, 0, stream>>>(cls0, pos, Wq, bq, q);
  k_logits  <<<(BATCH * NH * TP1) / 256, 256, 0, stream>>>(q, kv, lg);
  k_softmax <<<BATCH * NH, 256, 0, stream>>>(lg);
  k_pooled  <<<(BATCH * NH * HD) / 256, 256, 0, stream>>>(lg, kv, pooled);
  k_attnsp  <<<BATCH, 256, 0, stream>>>(lg, asp, out_asp);
  k_oproj   <<<BATCH * CH, 256, 0, stream>>>(pooled, Wo, bo, o);
  k_sal     <<<(BATCH * TOK) / 256, 256, 0, stream>>>(x, sal);
  k_topk    <<<BATCH, 256, 0, stream>>>(sal, idxb, K);
  k_vfused  <<<(BATCH * CH) / 256, 256, 0, stream>>>(x, idxb, o, cls0, vf, K);
  k_g1      <<<BATCH * G1DIM, 256, 0, stream>>>(vf, Wg1, h1);
  k_g2      <<<BATCH * CH, 256, 0, stream>>>(h1, Wg2, bg2, gate);
  k_final   <<<BATCH * CH, 256, 0, stream>>>(feat, x, gate, asp, post_w, post_b, out, alpha);
}
